// VQVAE_35433480192437
// MI455X (gfx1250) — matrix operand order repaired and checked by reading, not compiled, not run on hardware
//
#include <hip/hip_runtime.h>
#include <hip/hip_bf16.h>
#include <math.h>

typedef __attribute__((ext_vector_type(16))) __bf16 v16bf;
typedef __attribute__((ext_vector_type(8)))  float  v8f;

__device__ __forceinline__ unsigned short f2bf(float f) {
    union { float f; unsigned int u; } c; c.f = f;
    unsigned int u = c.u;
    unsigned int r = u + 0x7fffu + ((u >> 16) & 1u);   // round-to-nearest-even
    if ((u & 0x7f800000u) == 0x7f800000u) r = u;       // inf/nan passthrough
    return (unsigned short)(r >> 16);
}

// ---------------------------------------------------------------------------
// Weight pre-pack: OIHW fp32 -> bf16 A-fragments in the CDNA5 16x32 layout.
// Fragment index = ((tile*kSteps + s)*32 + lane)*16 + i.  Lane/K mapping per
// ISA 7.12.2 (16-bit A): lanes 0-15: K {0..7},{16..23}; lanes 16-31: K {8..15},{24..31}.
// ---------------------------------------------------------------------------
__global__ void pack_weights(const float* __restrict__ w, unsigned short* __restrict__ dst,
                             int Cout, int Cin, int KH, int KW, int kSteps, int total) {
    int idx = blockIdx.x * 256 + threadIdx.x;
    if (idx >= total) return;
    int i    = idx & 15;
    int lane = (idx >> 4) & 31;
    int ts   = idx >> 9;
    int s    = ts % kSteps;
    int t    = ts / kSteps;
    int m    = t * 16 + (lane & 15);
    int j = i >> 1, h = i & 1, grp = lane >> 4;
    int k32 = (j < 4) ? (grp * 8 + 2 * j + h)
                      : (16 + grp * 8 + 2 * (j - 4) + h);
    int kk = s * 32 + k32;
    int Kred = Cin * KH * KW;
    float v = 0.f;
    if (m < Cout && kk < Kred) {
        int cin = kk / (KH * KW);
        int r   = kk - cin * KH * KW;
        int kh  = r / KW, kw = r - (r / KW) * KW;
        v = w[((long)(m * Cin + cin) * KH + kh) * KW + kw];
    }
    dst[idx] = f2bf(v);
}

__global__ void emb_sq(const float* __restrict__ emb, float* __restrict__ e2) {
    int c = blockIdx.x * 256 + threadIdx.x;
    if (c >= 512) return;
    float s = 0.f;
    for (int d = 0; d < 64; ++d) { float v = emb[c * 64 + d]; s += v * v; }
    e2[c] = s;
}

// ---------------------------------------------------------------------------
// Implicit-GEMM conv. One wave32 block computes a 32(outch) x 16(pixel) tile:
// 2 A-fragments share one gathered B-fragment -> 2 WMMAs per k-step.
// Grid: (OW/16, B*OH, ceil(Cout/32)).  Requires OW % 16 == 0 (true for all
// convs in this net), so a pixel tile never crosses a row: no per-pixel
// divisions, row-validity check hoisted out of the gather loop.
// Fusions: IR relu-on-input, bias, RS residual add, OA (0/relu/tanh),
// SH pixel-shuffle(r=2) scatter store.
// ---------------------------------------------------------------------------
struct ConvParams {
    const float* in;
    const float* res;
    const float* bias;
    const unsigned short* wpk;
    float* out;
    int Cin, H, W;
    int Cout, KH, KW, stride, pad;
    int OH, OW;
    int Kred, kSteps, tiles;   // tiles = ceil(Cout/16)
};

template<int IR, int OA, int SH, int RS>
__global__ __launch_bounds__(32) void conv_wmma(ConvParams p) {
    const int lane = threadIdx.x;
    const int t0   = blockIdx.z * 2;                     // first 16-outch tile
    const int t1   = (t0 + 1 < p.tiles) ? (t0 + 1) : t0; // clamp for odd tile counts
    const int oy   = (int)blockIdx.y % p.OH;             // one scalar div per block
    const int b    = (int)blockIdx.y / p.OH;
    const int ox0  = (int)blockIdx.x * 16;

    __shared__ __align__(32) unsigned short Bt[512];     // 32x16 bf16 B tile, fragment order

    const int  iyRow = oy * p.stride - p.pad;
    const int  ixCol = ox0 * p.stride - p.pad;
    const long inB   = (long)b * p.Cin * p.H * p.W;
    const int  KHW   = p.KH * p.KW;
    const int  fragRow = (lane & 8) << 1;
    const int  fragCol = (lane & 7) | ((lane >> 1) & 8);

    const unsigned short* wA0 = p.wpk + (long)t0 * p.kSteps * 512 + lane * 16;
    const unsigned short* wA1 = p.wpk + (long)t1 * p.kSteps * 512 + lane * 16;

    v8f acc0 = {}, acc1 = {};
    for (int s = 0; s < p.kSteps; ++s) {
        // ---- gather: this lane supplies reduction-row k = s*32+lane --------
        int  kk  = s * 32 + lane;
        bool kv  = kk < p.Kred;
        int  cin = kk / KHW;
        int  r   = kk - cin * KHW;
        int  kh  = r / p.KW, kw = r - (r / p.KW) * p.KW;
        int  iy  = iyRow + kh;
        bool rowv = kv && iy >= 0 && iy < p.H;
        const float* rowPtr = p.in + inB + (long)cin * p.H * p.W + (long)iy * p.W;
#pragma unroll
        for (int n = 0; n < 16; ++n) {
            int ix = ixCol + n * p.stride + kw;
            float v = 0.f;
            if (rowv && ix >= 0 && ix < p.W) {
                v = rowPtr[ix];
                if (IR) v = fmaxf(v, 0.f);
            }
            Bt[(fragRow + n) * 16 + fragCol] = f2bf(v);
        }
        __syncthreads();
        if (s + 1 < p.kSteps) {                       // gfx1250 global_prefetch_b8
            __builtin_prefetch(wA0 + (long)(s + 1) * 512, 0, 1);
            __builtin_prefetch(wA1 + (long)(s + 1) * 512, 0, 1);
        }
        v16bf a0 = *(const v16bf*)(wA0 + (long)s * 512);
        v16bf a1 = *(const v16bf*)(wA1 + (long)s * 512);
        v16bf bf = *(const v16bf*)&Bt[lane * 16];
        acc0 = __builtin_amdgcn_wmma_f32_16x16x32_bf16(false, a0, false, bf, (short)0, acc0, false, false);
        acc1 = __builtin_amdgcn_wmma_f32_16x16x32_bf16(false, a1, false, bf, (short)0, acc1, false, false);
        __syncthreads();
    }

    // ---- epilogue: bias + residual + activation + (shuffled) store --------
    const int n  = lane & 15;
    const int mb = (lane & 16) >> 1;    // 0 or 8
    const int ox = ox0 + n;
#pragma unroll
    for (int half = 0; half < 2; ++half) {
        v8f acc = half ? acc1 : acc0;
        int ocB = (half ? t1 : t0) * 16 + mb;
        if (half && t1 == t0) break;    // clamped duplicate tile
#pragma unroll
        for (int v = 0; v < 8; ++v) {
            int oc = ocB + v;
            if (oc < p.Cout) {
                float val = acc[v] + p.bias[oc];
                long oidx;
                if (SH) {
                    int c2 = oc >> 2, rr = (oc >> 1) & 1, ss = oc & 1;
                    oidx = (((long)b * (p.Cout >> 2) + c2) * (p.OH * 2) + (oy * 2 + rr))
                               * (long)(p.OW * 2) + (ox * 2 + ss);
                } else {
                    oidx = (((long)b * p.Cout + oc) * p.OH + oy) * p.OW + ox;
                    if (RS) val += p.res[oidx];
                }
                if (OA == 1)      val = fmaxf(val, 0.f);
                else if (OA == 2) val = tanhf(val);
                p.out[oidx] = val;
            }
        }
    }
}

// ---------------------------------------------------------------------------
// VQ: 16 pixels per wave; 32 code-tiles x (K=64 -> 2 WMMA steps) of z.e^T;
// dist = |z|^2 + |e|^2 - 2 z.e ; argmin ; gather z_q ; loss/count atomics.
// Grid: B * HW/16 blocks (HW % 16 == 0).
// ---------------------------------------------------------------------------
__global__ __launch_bounds__(32) void vq_wmma(const float* __restrict__ ze,
                                              const unsigned short* __restrict__ epk,
                                              const float* __restrict__ e2,
                                              const float* __restrict__ embf,
                                              float* __restrict__ zq,
                                              float* __restrict__ counts,
                                              float* __restrict__ lossAcc,
                                              int HW, int HW16) {
    const int lane = threadIdx.x;
    const int b    = (int)blockIdx.x / HW16;     // one scalar div per block
    const int r0   = ((int)blockIdx.x - b * HW16) * 16;
    __shared__ __align__(32) unsigned short Bt[1024];
    __shared__ float z2s[16];

    const long zBase = (long)b * 64 * HW + r0;   // + d*HW + n
    const int fragRow = (lane & 8) << 1;
    const int fragCol = (lane & 7) | ((lane >> 1) & 8);
#pragma unroll
    for (int s = 0; s < 2; ++s) {
        int d = s * 32 + lane;
        const float* zp = ze + zBase + (long)d * HW;
#pragma unroll
        for (int n = 0; n < 16; ++n)
            Bt[s * 512 + (fragRow + n) * 16 + fragCol] = f2bf(zp[n]);
    }
    if (lane < 16) {
        float s2 = 0.f;
        const float* zp = ze + zBase + lane;
        for (int d = 0; d < 64; ++d) { float z = zp[(long)d * HW]; s2 += z * z; }
        z2s[lane] = s2;
    }
    __syncthreads();

    v16bf b0 = *(const v16bf*)&Bt[lane * 16];
    v16bf b1 = *(const v16bf*)&Bt[512 + lane * 16];
    float z2n = z2s[lane & 15];
    float bestd = 3.4e38f;
    int   besti = 0;
    const int mb = (lane & 16) >> 1;

    for (int ct = 0; ct < 32; ++ct) {
        v8f acc = {};
        v16bf a0 = *(const v16bf*)(epk + (((long)ct * 2 + 0) * 32 + lane) * 16);
        v16bf a1 = *(const v16bf*)(epk + (((long)ct * 2 + 1) * 32 + lane) * 16);
        acc = __builtin_amdgcn_wmma_f32_16x16x32_bf16(false, a0, false, b0, (short)0, acc, false, false);
        acc = __builtin_amdgcn_wmma_f32_16x16x32_bf16(false, a1, false, b1, (short)0, acc, false, false);
#pragma unroll
        for (int v = 0; v < 8; ++v) {
            int code = ct * 16 + mb + v;
            float dst = z2n + e2[code] - 2.0f * acc[v];
            if (dst < bestd) { bestd = dst; besti = code; }
        }
    }
    // combine lane L (codes 0-7 of each tile) with L+16 (codes 8-15)
    float od = __shfl_xor(bestd, 16, 32);
    int   oi = __shfl_xor(besti, 16, 32);
    if (od < bestd || (od == bestd && oi < besti)) { bestd = od; besti = oi; }

    if (lane < 16) {
        const float* ep = embf + (long)besti * 64;
        float sq = 0.f;
        const float* zp = ze + zBase + lane;
        float*       qp = zq + zBase + lane;
        for (int d = 0; d < 64; ++d) {
            float e = ep[d];
            float z = zp[(long)d * HW];
            float df = e - z;
            sq += df * df;
            qp[(long)d * HW] = e;                // straight-through forward == z_q
        }
        atomicAdd(lossAcc, sq);
        atomicAdd(&counts[besti], 1.0f);
    }
}

__global__ void vq_finalize(const float* __restrict__ counts, const float* __restrict__ lossAcc,
                            float* __restrict__ out, long outOff, int Npix) {
    __shared__ float red[256];
    int t = threadIdx.x;
    float part = 0.f;
    for (int c = t; c < 512; c += 256) {
        float em = counts[c] / (float)Npix;
        part += em * logf(em + 1e-10f);
    }
    red[t] = part;
    __syncthreads();
    for (int s = 128; s > 0; s >>= 1) {
        if (t < s) red[t] += red[t + s];
        __syncthreads();
    }
    if (t == 0) {
        out[outOff]     = 1.25f * (*lossAcc) / ((float)Npix * 64.0f);  // (1+BETA)*mean
        out[outOff + 1] = expf(-red[0]);
    }
}

// ---------------------------------------------------------------------------
// Host side
// ---------------------------------------------------------------------------
static void launch_conv(const float* in, const float* res, const float* bias,
                        const unsigned short* wpk, float* out,
                        int B, int Cin, int H, int W, int Cout, int KH, int KW,
                        int stride, int pad, int inRelu, int outAct, int shuffle,
                        hipStream_t st) {
    ConvParams p;
    p.in = in; p.res = res; p.bias = bias; p.wpk = wpk; p.out = out;
    p.Cin = Cin; p.H = H; p.W = W;
    p.Cout = Cout; p.KH = KH; p.KW = KW; p.stride = stride; p.pad = pad;
    p.OH = (H + 2 * pad - KH) / stride + 1;
    p.OW = (W + 2 * pad - KW) / stride + 1;
    p.Kred = Cin * KH * KW;
    p.kSteps = (p.Kred + 31) / 32;
    p.tiles = (Cout + 15) / 16;
    dim3 grid(p.OW / 16, B * p.OH, (Cout + 31) / 32);
    bool rs = (res != nullptr);
    if      (inRelu == 0 && outAct == 0 && !shuffle && !rs) conv_wmma<0,0,0,0><<<grid,32,0,st>>>(p);
    else if (inRelu == 1 && outAct == 0 && !shuffle && !rs) conv_wmma<1,0,0,0><<<grid,32,0,st>>>(p);
    else if (inRelu == 1 && outAct == 0 && !shuffle &&  rs) conv_wmma<1,0,0,1><<<grid,32,0,st>>>(p);
    else if (inRelu == 1 && outAct == 1 &&  shuffle && !rs) conv_wmma<1,1,1,0><<<grid,32,0,st>>>(p);
    else                                                    conv_wmma<0,2,1,0><<<grid,32,0,st>>>(p);
}

extern "C" void kernel_launch(void* const* d_in, const int* in_sizes, int n_in,
                              void* d_out, int out_size, void* d_ws, size_t ws_size,
                              hipStream_t stream) {
    (void)in_sizes; (void)n_in; (void)out_size; (void)ws_size;
    const float* x     = (const float*)d_in[0];
    const float* ec0_w = (const float*)d_in[1];
    const float* ec0_b = (const float*)d_in[2];
    const float* ec1_w = (const float*)d_in[3];
    const float* ec1_b = (const float*)d_in[4];
    const float* erw1  = (const float*)d_in[5];
    const float* erb1  = (const float*)d_in[6];
    const float* erw2  = (const float*)d_in[7];
    const float* erb2  = (const float*)d_in[8];
    const float* pq_w  = (const float*)d_in[9];
    const float* pq_b  = (const float*)d_in[10];
    const float* emb   = (const float*)d_in[11];
    const float* dc0_w = (const float*)d_in[12];
    const float* dc0_b = (const float*)d_in[13];
    const float* drw1  = (const float*)d_in[14];
    const float* drb1  = (const float*)d_in[15];
    const float* drw2  = (const float*)d_in[16];
    const float* drb2  = (const float*)d_in[17];
    const float* up1_w = (const float*)d_in[18];
    const float* up1_b = (const float*)d_in[19];
    const float* up0_w = (const float*)d_in[20];
    const float* up0_b = (const float*)d_in[21];
    float* out = (float*)d_out;

    // ---- workspace layout ----
    unsigned short* wp = (unsigned short*)d_ws;     // packed weights at offset 0 (~2 MiB)
    size_t off = 0;                                  // in halves
    auto pack = [&](const float* w, int Cout, int Cin, int KH, int KW) -> size_t {
        int tiles = (Cout + 15) / 16;
        int kSteps = (Cin * KH * KW + 31) / 32;
        int total = tiles * kSteps * 512;
        size_t my = off;
        pack_weights<<<(total + 255) / 256, 256, 0, stream>>>(w, wp + my, Cout, Cin, KH, KW, kSteps, total);
        off += (size_t)total;
        return my;
    };
    size_t o_ec0 = pack(ec0_w, 64, 3, 4, 4);
    size_t o_ec1 = pack(ec1_w, 128, 64, 4, 4);
    size_t o_er1 = pack(erw1, 64, 128, 3, 3);
    size_t o_er2 = pack(erw2, 128, 64, 1, 1);
    size_t o_pq  = pack(pq_w, 64, 128, 1, 1);
    size_t o_emb = pack(emb, 512, 64, 1, 1);
    size_t o_dc0 = pack(dc0_w, 128, 64, 3, 3);
    size_t o_dr1 = pack(drw1, 64, 128, 3, 3);
    size_t o_dr2 = pack(drw2, 128, 64, 1, 1);
    size_t o_up1 = pack(up1_w, 512, 128, 3, 3);
    size_t o_up0 = pack(up0_w, 12, 128, 3, 3);

    // scalar region at 4 MiB
    char* base = (char*)d_ws;
    float* e2      = (float*)(base + (4u << 20));
    float* counts  = e2 + 512;
    float* lossAcc = counts + 512;
    // activation arena at 8 MiB (MiB offsets, with reuse)
    float* A   = (float*)(base + (8u << 20));
    float* a0  = A;                                   // [0,64)   enc_c0 out  16x64x128x128
    float* a1  = (float*)((char*)A + (64u  << 20));   // [64,96)  enc_c1 / res x
    float* rh  = (float*)((char*)A + (96u  << 20));   // [96,112) res temp 16x64x64x64
    float* ze  = A;                                   // [0,16)   reuse (a0 dead)
    float* zqb = (float*)((char*)A + (16u  << 20));   // [16,32)
    float* d0  = (float*)((char*)A + (32u  << 20));   // [32,64)  dec_c0 out / dec x
    float* sh1 = (float*)((char*)A + (112u << 20));   // [112,240) shuffled up1 out

    emb_sq<<<2, 256, 0, stream>>>(emb, e2);
    hipMemsetAsync(counts, 0, (512 + 1) * sizeof(float), stream);   // counts + lossAcc

    // ---------------- encoder ----------------
    launch_conv(x,  nullptr, ec0_b, wp + o_ec0, a0, 16, 3,   256, 256, 64,  4, 4, 2, 1, 0, 0, 0, stream);
    launch_conv(a0, nullptr, ec1_b, wp + o_ec1, a1, 16, 64,  128, 128, 128, 4, 4, 2, 1, 1, 0, 0, stream);
    for (int it = 0; it < 2; ++it) {
        launch_conv(a1, nullptr, erb1, wp + o_er1, rh, 16, 128, 64, 64, 64,  3, 3, 1, 1, 1, 0, 0, stream);
        launch_conv(rh, a1,      erb2, wp + o_er2, a1, 16, 64,  64, 64, 128, 1, 1, 1, 0, 1, 0, 0, stream);
    }
    launch_conv(a1, nullptr, pq_b, wp + o_pq, ze, 16, 128, 64, 64, 64, 1, 1, 1, 0, 1, 0, 0, stream);

    // ---------------- vector quantization ----------------
    {
        int HW = 64 * 64, HW16 = HW / 16;
        int totalPix = 16 * HW;
        vq_wmma<<<totalPix / 16, 32, 0, stream>>>(ze, wp + o_emb, e2, emb, zqb, counts, lossAcc, HW, HW16);
        vq_finalize<<<1, 256, 0, stream>>>(counts, lossAcc, out, (long)16 * 3 * 256 * 256, totalPix);
    }

    // ---------------- decoder ----------------
    launch_conv(zqb, nullptr, dc0_b, wp + o_dc0, d0, 16, 64, 64, 64, 128, 3, 3, 1, 1, 0, 0, 0, stream);
    for (int it = 0; it < 2; ++it) {
        launch_conv(d0, nullptr, drb1, wp + o_dr1, rh, 16, 128, 64, 64, 64,  3, 3, 1, 1, 1, 0, 0, stream);
        launch_conv(rh, d0,      drb2, wp + o_dr2, d0, 16, 64,  64, 64, 128, 1, 1, 1, 0, 1, 0, 0, stream);
    }
    launch_conv(d0, nullptr, up1_b, wp + o_up1, sh1, 16, 128, 64, 64, 512, 3, 3, 1, 1, 1, 1, 1, stream);
    launch_conv(sh1, nullptr, up0_b, wp + o_up0, out, 16, 128, 128, 128, 12, 3, 3, 1, 1, 0, 2, 1, stream);
}
